// ImageRegistration_45260365365522
// MI455X (gfx1250) — matrix operand order repaired and checked by reading, not compiled, not run on hardware
//
#include <hip/hip_runtime.h>
#include <hip/hip_bf16.h>
#include <math.h>

typedef __attribute__((ext_vector_type(16))) _Float16 v16h;
typedef __attribute__((ext_vector_type(8)))  _Float16 v8h;
typedef __attribute__((ext_vector_type(8)))  float    v8f;

#define NEG_INF  (-1000000000.0f)
#define LN2048   7.6246189861593985f
#define LN1024   6.9314718055994531f
#define FSCALE   0.17677669529663687f   /* 32^-0.5 per side -> mm carries 1/32 */

static constexpr int    NB = 2, L = 1024, EC = 8, CC = 32, KK = 256, LP = 1025;

/* d_out element offsets (fp32) */
static constexpr size_t S0      = (size_t)NB * LP * LP;          /* score_map   */
static constexpr size_t ANG_OFF = S0;                            /* angle_map x2*/
static constexpr size_t SCL_OFF = ANG_OFF + (size_t)NB * L * L * 2;
static constexpr size_t TRN_OFF = SCL_OFF + (size_t)NB * L * L;

/* workspace byte offsets */
static constexpr size_t ALGN(size_t x) { return (x + 255) & ~(size_t)255; }
static constexpr size_t F16_SZ   = (size_t)NB * L * KK * 2;      /* 1 MB each   */
static constexpr size_t FI16_OFF = 0;
static constexpr size_t FT16_OFF = FI16_OFF + F16_SZ;
static constexpr size_t CW_OFF   = ALGN(FT16_OFF + F16_SZ);      /* couplings   */
static constexpr size_t CW_SZ    = (size_t)NB * LP * LP * 4;
static constexpr size_t U_OFF    = ALGN(CW_OFF + CW_SZ);
static constexpr size_t V_OFF    = ALGN(U_OFF + (size_t)NB * LP * 4);
static constexpr size_t WB_OFF   = ALGN(V_OFF + (size_t)NB * LP * 4);

/* ---- collapse the linear MLP: Weff = W1@W2@W3 (8x6), beff (6) ---- */
__global__ __launch_bounds__(64) void prep_weights(
    const float* __restrict__ W1, const float* __restrict__ b1,
    const float* __restrict__ W2, const float* __restrict__ b2,
    const float* __restrict__ W3, const float* __restrict__ b3,
    float* __restrict__ wb)
{
  __shared__ float t1[8 * 64];
  __shared__ float bb[64];
  int t = threadIdx.x;
  for (int idx = t; idx < 512; idx += 64) {
    int e = idx >> 6, h2 = idx & 63;
    float acc = 0.f;
    for (int h1 = 0; h1 < 64; ++h1) acc += W1[e * 64 + h1] * W2[h1 * 64 + h2];
    t1[idx] = acc;
  }
  {
    float acc = b2[t];
    for (int h1 = 0; h1 < 64; ++h1) acc += b1[h1] * W2[h1 * 64 + t];
    bb[t] = acc;
  }
  __syncthreads();
  if (t < 48) {
    int e = t / 6, j = t % 6;
    float acc = 0.f;
    for (int h2 = 0; h2 < 64; ++h2) acc += t1[e * 64 + h2] * W3[h2 * 6 + j];
    wb[t] = acc;                                   /* Weff[e][j] at e*6+j */
  }
  if (t < 6) {
    float acc = b3[t];
    for (int h2 = 0; h2 < 64; ++h2) acc += bb[h2] * W3[h2 * 6 + t];
    wb[48 + t] = acc;                              /* beff[j] */
  }
}

/* ---- fp32 features -> pre-scaled f16 ---- */
__global__ __launch_bounds__(256) void convert_f16(
    const float* __restrict__ fi, const float* __restrict__ ft,
    _Float16* __restrict__ fi16, _Float16* __restrict__ ft16, int n)
{
  int i = blockIdx.x * 256 + threadIdx.x;
  if (i < n) {
    fi16[i] = (_Float16)(fi[i] * FSCALE);
    ft16[i] = (_Float16)(ft[i] * FSCALE);
  }
}

/* ---- dustbin row/col of couplings = alpha; u=v=0 ---- */
__global__ __launch_bounds__(256) void fill_bins(
    float* __restrict__ cw, float* __restrict__ u, float* __restrict__ v,
    const float* __restrict__ alpha_p)
{
  int i = blockIdx.x * 256 + threadIdx.x;
  if (i >= NB * LP) return;
  int b = i / LP, k = i % LP;
  float a = *alpha_p;
  cw[((size_t)b * LP + (LP - 1)) * LP + k] = a;
  cw[((size_t)b * LP + k) * LP + (LP - 1)] = a;
  u[i] = 0.f;
  v[i] = 0.f;
}

/* ---- fused WMMA contraction + Weff combine + head outputs ----
 * One wave32 per 16x16 (l,s) tile. 8 e-chunks of K=32 -> 8 WMMAs, 8 f32 accs.
 * A(16x32 f16): lane<16 holds M=lane, K0..15 contiguous; lane>=16 K16..31
 * B(32x16 f16): lane<16 holds N=lane, K0..15 contiguous; lane>=16 K16..31
 * C(16x16 f32): vgpr r, lane L -> M = r + 8*(L>=16), N = L&15
 */
union AFrag { v16h v; v8h h[2]; };

__global__ __launch_bounds__(256) void wmma_m6_kernel(
    const _Float16* __restrict__ fi16, const _Float16* __restrict__ ft16,
    const float* __restrict__ wb,
    const unsigned char* __restrict__ mi, const unsigned char* __restrict__ mt,
    float* __restrict__ cw, float* __restrict__ out)
{
  int wave = (blockIdx.x * 256 + threadIdx.x) >> 5;   /* 0..8191 */
  int lane = threadIdx.x & 31;
  int n    = wave >> 12;
  int rem  = wave & 4095;
  int l0   = (rem >> 6) << 4;
  int s0   = (rem & 63) << 4;
  int half = lane >> 4;
  int mrow = lane & 15;

  const _Float16* aBase = fi16 + ((size_t)(n * L + l0 + mrow)) * KK + half * 16;
  const _Float16* bBase = ft16 + ((size_t)(n * L + s0 + mrow)) * KK + half * 16;

  v8f acc[8] = {};
#pragma unroll
  for (int e = 0; e < 8; ++e) {
    AFrag a;
    a.h[0] = *(const v8h*)(aBase + e * 32);
    a.h[1] = *(const v8h*)(aBase + e * 32 + 8);
    v16h b = *(const v16h*)(bBase + e * 32);
    acc[e] = __builtin_amdgcn_wmma_f32_16x16x32_f16(
        false, a.v, false, b, (short)0, acc[e], false, false);
  }

  /* combine over e with Weff (VALU, co-executes with WMMA pipeline) */
  v8f m6c[6];
#pragma unroll
  for (int j = 0; j < 6; ++j) {
    v8f o = {};
    float bj = wb[48 + j];
#pragma unroll
    for (int i = 0; i < 8; ++i) o[i] = bj;
#pragma unroll
    for (int e = 0; e < 8; ++e) o += acc[e] * wb[e * 6 + j];
    m6c[j] = o;
  }

  int s = s0 + mrow;
  unsigned char mts = mt[n * L + s];
#pragma unroll
  for (int r = 0; r < 8; ++r) {
    int l = l0 + half * 8 + r;
    unsigned char mis = mi[n * L + l];
    size_t pidx = ((size_t)n * L + l) * L + s;

    float sc = (mis && mts) ? m6c[0][r] : NEG_INF;
    cw[((size_t)n * LP + l) * LP + s] = sc;

    float a0 = m6c[1][r], a1 = m6c[2][r];
    float d = fmaxf(sqrtf(a0 * a0 + a1 * a1), 1e-12f);
    *(float2*)(out + ANG_OFF + pidx * 2) = make_float2(a0 / d, a1 / d);
    out[SCL_OFF + pidx] = m6c[3][r];
    *(float2*)(out + TRN_OFF + pidx * 2) = make_float2(m6c[4][r], m6c[5][r]);
  }
}

/* ---- Sinkhorn: u = log_mu - LSE_j(cw + v) ---- */
__global__ __launch_bounds__(256) void row_lse(
    const float* __restrict__ cw, const float* __restrict__ v,
    float* __restrict__ u)
{
  int row = blockIdx.x;               /* 0..2049 */
  int b = row / LP, i = row % LP;
  const float* rp = cw + ((size_t)b * LP + i) * LP;
  const float* vp = v + b * LP;
  float m = -INFINITY, sacc = 0.f;
  for (int j = threadIdx.x; j < LP; j += 256) {
    float x = rp[j] + vp[j];
    float mn = fmaxf(m, x);
    sacc = sacc * __expf(m - mn) + __expf(x - mn);
    m = mn;
  }
  __shared__ float sm[256], ss[256];
  sm[threadIdx.x] = m; ss[threadIdx.x] = sacc;
  __syncthreads();
  for (int o = 128; o > 0; o >>= 1) {
    if (threadIdx.x < o) {
      float m2 = sm[threadIdx.x + o], s2 = ss[threadIdx.x + o];
      float mn = fmaxf(sm[threadIdx.x], m2);
      ss[threadIdx.x] = ss[threadIdx.x] * __expf(sm[threadIdx.x] - mn)
                        + s2 * __expf(m2 - mn);
      sm[threadIdx.x] = mn;
    }
    __syncthreads();
  }
  if (threadIdx.x == 0) {
    float lmu = (i < L) ? -LN2048 : (LN1024 - LN2048);
    u[b * LP + i] = lmu - (sm[0] + __logf(ss[0]));
  }
}

/* ---- Sinkhorn: v = log_nu - LSE_i(cw + u), one column per thread ---- */
__global__ __launch_bounds__(256) void col_lse(
    const float* __restrict__ cw, const float* __restrict__ u,
    float* __restrict__ v)
{
  int b = blockIdx.x / 5;
  int j = (blockIdx.x % 5) * 256 + threadIdx.x;
  if (j >= LP) return;
  const float* up = u + b * LP;
  const float* cp = cw + (size_t)b * LP * LP + j;
  float m = -INFINITY, sacc = 0.f;
  for (int i = 0; i < LP; ++i) {
    float x = cp[(size_t)i * LP] + up[i];
    float mn = fmaxf(m, x);
    sacc = sacc * __expf(m - mn) + __expf(x - mn);
    m = mn;
  }
  float lnu = (j < L) ? -LN2048 : (LN1024 - LN2048);
  v[b * LP + j] = lnu - (m + __logf(sacc));
}

/* ---- score_map = exp(cw + u + v - norm) ---- */
__global__ __launch_bounds__(256) void final_exp(
    const float* __restrict__ cw, const float* __restrict__ u,
    const float* __restrict__ v, float* __restrict__ out)
{
  size_t i = (size_t)blockIdx.x * 256 + threadIdx.x;
  if (i >= S0) return;
  int b = (int)(i / ((size_t)LP * LP));
  int rem = (int)(i % ((size_t)LP * LP));
  int r = rem / LP, c = rem % LP;
  out[i] = __expf(cw[i] + u[b * LP + r] + v[b * LP + c] + LN2048);
}

extern "C" void kernel_launch(void* const* d_in, const int* in_sizes, int n_in,
                              void* d_out, int out_size, void* d_ws, size_t ws_size,
                              hipStream_t stream) {
  const float* feat_init = (const float*)d_in[0];
  const float* feat_term = (const float*)d_in[1];
  const float* W1 = (const float*)d_in[2];
  const float* b1 = (const float*)d_in[3];
  const float* W2 = (const float*)d_in[4];
  const float* b2 = (const float*)d_in[5];
  const float* W3 = (const float*)d_in[6];
  const float* b3 = (const float*)d_in[7];
  const float* bin_score = (const float*)d_in[8];
  const unsigned char* mask_init = (const unsigned char*)d_in[9];
  const unsigned char* mask_term = (const unsigned char*)d_in[10];
  float* out = (float*)d_out;

  char* ws = (char*)d_ws;
  _Float16* fi16 = (_Float16*)(ws + FI16_OFF);
  _Float16* ft16 = (_Float16*)(ws + FT16_OFF);
  float* cw = (float*)(ws + CW_OFF);
  float* u  = (float*)(ws + U_OFF);
  float* v  = (float*)(ws + V_OFF);
  float* wb = (float*)(ws + WB_OFF);

  prep_weights<<<1, 64, 0, stream>>>(W1, b1, W2, b2, W3, b3, wb);

  int nElem = NB * L * EC * CC;                 /* 524288 */
  convert_f16<<<(nElem + 255) / 256, 256, 0, stream>>>(feat_init, feat_term,
                                                       fi16, ft16, nElem);

  fill_bins<<<(NB * LP + 255) / 256, 256, 0, stream>>>(cw, u, v, bin_score);

  /* 8192 tiles of 16x16, 8 waves per block */
  wmma_m6_kernel<<<1024, 256, 0, stream>>>(fi16, ft16, wb, mask_init, mask_term,
                                           cw, out);

  for (int it = 0; it < 3; ++it) {
    row_lse<<<NB * LP, 256, 0, stream>>>(cw, v, u);
    col_lse<<<NB * 5, 256, 0, stream>>>(cw, u, v);
  }

  final_exp<<<(int)((S0 + 255) / 256), 256, 0, stream>>>(cw, u, v, out);
}
